// FlatLIFSNN_54683523612727
// MI455X (gfx1250) — matrix operand order repaired and checked by reading, not compiled, not run on hardware
//
#include <hip/hip_runtime.h>
#include <hip/hip_fp16.h>

// FlatLIF SNN for MI455X (gfx1250, wave32, WMMA).
// Memory-bound (252MB input @ 23.3TB/s ~11us floor). All large GEMMs use
// v_wmma_f32_16x16x32_f16 (f32 accumulate); activations carried in f16 to
// halve intermediate HBM traffic; A-tiles staged to LDS via the CDNA5
// async global->LDS path (ASYNCcnt); LIF scans stream with global_prefetch.

typedef __attribute__((ext_vector_type(16))) _Float16 v16h;
typedef __attribute__((ext_vector_type(8)))  float    v8f;

#define LIF_BETA 0.95f
#define LIF_THRESH 1.0f
#define LN_EPS 1e-5f

union Frag16 { v16h v; uint4 u[2]; };

__device__ __forceinline__ v8f wmma16x16x32(v16h a, v16h b, v8f c) {
    // (neg_a, A, neg_b, B, c_mod, C, reuse_a, reuse_b)
    return __builtin_amdgcn_wmma_f32_16x16x32_f16(false, a, false, b,
                                                  (short)0, c, false, false);
}

// Async global->LDS staging (CDNA5, ASYNCcnt-tracked) with sync fallback.
// Probe round 2: builtin exists, 4 args, param0 = v4i32 in AS(1) (non-const).
#if defined(__gfx1250__) && __has_builtin(__builtin_amdgcn_global_load_async_to_lds_b128)
#define USE_ASYNC_LDS 1
typedef int v4i_raw __attribute__((vector_size(16)));
typedef __attribute__((address_space(1))) v4i_raw gbl_v4i;
typedef __attribute__((address_space(3))) v4i_raw lds_v4i;
#else
#define USE_ASYNC_LDS 0
#endif

__device__ __forceinline__ void wait_async0() {
#if __has_builtin(__builtin_amdgcn_s_wait_asynccnt)
    __builtin_amdgcn_s_wait_asynccnt(0);
#else
    asm volatile("s_wait_asynccnt 0x0" ::: "memory");
#endif
}

// ---------------------------------------------------------------- weights f32->f16
__global__ void cvt_f32_to_f16(const float* __restrict__ src,
                               _Float16* __restrict__ dst, int n) {
    int i = blockIdx.x * blockDim.x + threadIdx.x;
    if (i < n) dst[i] = (_Float16)src[i];
}

// ---------------------------------------------------------------- GEMM1 + LN + tanh
// x: [256,10,512,48] gathered as rows [b*512+t][c*48+f]; Wh: [128,480] f16.
// Block = 16 rows x 128 cols, 8 waves, each wave one 16x16 tile, 15 WMMAs (K=480).
__global__ __launch_bounds__(256) void gemm_in_ln(
    const float* __restrict__ x, const _Float16* __restrict__ Wh,
    const float* __restrict__ bias, const float* __restrict__ gamma,
    const float* __restrict__ beta, _Float16* __restrict__ Hout)
{
    __shared__ __align__(16) _Float16 xa[16 * 480];   // A tile, row-major f16
    __shared__ __align__(16) float    lnb[16 * 132];  // C slab for LayerNorm

    const int tid = threadIdx.x;
    const int m0  = blockIdx.x * 16;

    // Cooperative transpose-gather, vectorized: 1920 float4 (48 % 4 == 0, so a
    // float4 never crosses a channel boundary). 192B-contiguous runs per (row,c).
    for (int q = tid; q < 16 * 120; q += 256) {
        int row = q / 120, d4 = q - row * 120;
        int c = d4 / 12, f4 = d4 - c * 12;
        int rg = m0 + row;
        int b = rg >> 9, t = rg & 511;               // T = 512
        float4 v = *(const float4*)&x[(((b * 10 + c) << 9) + t) * 48 + f4 * 4];
        union { _Float16 h[4]; uint2 u; } o;
        o.h[0] = (_Float16)v.x; o.h[1] = (_Float16)v.y;
        o.h[2] = (_Float16)v.z; o.h[3] = (_Float16)v.w;
        *(uint2*)(xa + row * 480 + c * 48 + f4 * 4) = o.u;
    }
    __syncthreads();

    const int lane = tid & 31;
    const int wid  = tid >> 5;       // 0..7 -> 16-col tile
    const int n0   = wid * 16;
    const int ar   = lane & 15;      // A row / B column
    const int hi   = lane >> 4;      // K-half select (ISA 16-bit A/B layout)

    v8f acc = {};
    for (int kb = 0; kb < 480; kb += 32) {
        Frag16 a, b;
        const _Float16* ap = &xa[ar * 480 + kb + hi * 8];
        a.u[0] = *(const uint4*)(ap);
        a.u[1] = *(const uint4*)(ap + 16);
        const _Float16* bp = Wh + (n0 + ar) * 480 + kb + hi * 8;
        b.u[0] = *(const uint4*)(bp);
        b.u[1] = *(const uint4*)(bp + 16);
        acc = wmma16x16x32(a.v, b.v, acc);
    }

    // C layout: lane column = n0+ar, VGPR r -> row r + 8*hi. Add bias, park in LDS.
    float bv = bias[n0 + ar];
#pragma unroll
    for (int r = 0; r < 8; ++r)
        lnb[(r + 8 * hi) * 132 + n0 + ar] = acc[r] + bv;
    __syncthreads();

    // LayerNorm over h=128 + tanh. 16 threads per row, 8 cols each,
    // 16-lane shuffle reduction (wave32 -> two row-groups per wave).
    int row = tid >> 4, seg = tid & 15;
    float vloc[8], s = 0.f, sq = 0.f;
#pragma unroll
    for (int j = 0; j < 8; ++j) {
        float v = lnb[row * 132 + seg * 8 + j];
        vloc[j] = v; s += v; sq += v * v;
    }
#pragma unroll
    for (int off = 1; off < 16; off <<= 1) {
        s  += __shfl_xor(s,  off, 16);
        sq += __shfl_xor(sq, off, 16);
    }
    float mu   = s * (1.f / 128.f);
    float var  = sq * (1.f / 128.f) - mu * mu;
    float rstd = rsqrtf(var + LN_EPS);

    union { _Float16 h[8]; uint4 u; } o;
#pragma unroll
    for (int j = 0; j < 8; ++j) {
        int c = seg * 8 + j;
        float y = (vloc[j] - mu) * rstd * gamma[c] + beta[c];
        o.h[j] = (_Float16)tanhf(y);
    }
    *(uint4*)(Hout + (size_t)(m0 + row) * 128 + seg * 8) = o.u;
}

// ---------------------------------------------------------------- hidden GEMM (K=128)
// cur[m][n] = A[m][:] . W[n][:] + bias[n]; A f16 [M,128], W f16 [128,128].
// Block = 32 rows x 128 cols, 8 waves, 2 row-tiles per wave -> each B-frag
// feeds 2 WMMAs (8 WMMAs/wave). A staged via async global->LDS when available.
__global__ __launch_bounds__(256) void gemm_hidden(
    const _Float16* __restrict__ A, const _Float16* __restrict__ Wh,
    const float* __restrict__ bias, float* __restrict__ out)
{
    __shared__ __align__(16) _Float16 at[32 * 128];   // 8 KB
    const int tid = threadIdx.x;
    const int m0  = blockIdx.x * 32;

    {   // stage 4096 halfs: 32B per thread
        int idx = tid * 16;
        const _Float16* gp = A + (size_t)m0 * 128 + idx;
#if USE_ASYNC_LDS
        // Integer-valued AS casts: generic LDS pointer low 32 bits = LDS address.
        gbl_v4i* gv = (gbl_v4i*)(uintptr_t)gp;
        lds_v4i* lv = (lds_v4i*)(uint32_t)(uintptr_t)(at + idx);
        __builtin_amdgcn_global_load_async_to_lds_b128(gv, lv, 0, 0);
        __builtin_amdgcn_global_load_async_to_lds_b128(gv, lv, 16, 0);
        wait_async0();
#else
        *(uint4*)(at + idx)     = *(const uint4*)(gp);
        *(uint4*)(at + idx + 8) = *(const uint4*)(gp + 8);
#endif
    }
    __syncthreads();

    const int lane = tid & 31, wid = tid >> 5, n0 = wid * 16;
    const int ar = lane & 15, hi = lane >> 4;

    v8f acc0 = {}, acc1 = {};
#pragma unroll
    for (int kb = 0; kb < 128; kb += 32) {
        Frag16 a0, a1, b;
        const _Float16* bp = Wh + (n0 + ar) * 128 + kb + hi * 8;
        b.u[0] = *(const uint4*)(bp);
        b.u[1] = *(const uint4*)(bp + 16);
        const _Float16* ap0 = at + ar * 128 + kb + hi * 8;
        a0.u[0] = *(const uint4*)(ap0);
        a0.u[1] = *(const uint4*)(ap0 + 16);
        const _Float16* ap1 = ap0 + 16 * 128;
        a1.u[0] = *(const uint4*)(ap1);
        a1.u[1] = *(const uint4*)(ap1 + 16);
        acc0 = wmma16x16x32(a0.v, b.v, acc0);
        acc1 = wmma16x16x32(a1.v, b.v, acc1);
    }

    float bv  = bias[n0 + ar];
    float* op = out + (size_t)m0 * 128 + n0 + ar;
#pragma unroll
    for (int r = 0; r < 8; ++r) {
        op[(size_t)(r + 8 * hi) * 128]        = acc0[r] + bv;
        op[(size_t)(16 + r + 8 * hi) * 128]   = acc1[r] + bv;
    }
}

// ---------------------------------------------------------------- LIF scans
// One thread per (b,h); consecutive h -> coalesced 512B lines per step.
__global__ void lif_spk(const float* __restrict__ cur, _Float16* __restrict__ spk) {
    int idx = blockIdx.x * blockDim.x + threadIdx.x;   // 32768 = 256*128
    int b = idx >> 7, h = idx & 127;
    const float* c = cur + ((size_t)b * 512) * 128 + h;
    _Float16*    s = spk + ((size_t)b * 512) * 128 + h;
    float mem = 0.f;
    for (int t = 0; t < 512; ++t) {
        if ((t & 3) == 0 && t + 32 < 512)
            __builtin_prefetch(&c[(size_t)(t + 32) * 128], 0, 1);
        mem = LIF_BETA * mem + c[(size_t)t * 128];
        float sv = (mem - LIF_THRESH) > 0.f ? 1.f : 0.f;
        s[(size_t)t * 128] = (_Float16)sv;
        mem -= sv * LIF_THRESH;
    }
}

__global__ void lif_mean(const float* __restrict__ cur, float* __restrict__ meanout) {
    int idx = blockIdx.x * blockDim.x + threadIdx.x;
    int b = idx >> 7, h = idx & 127;
    const float* c = cur + ((size_t)b * 512) * 128 + h;
    float mem = 0.f, sum = 0.f;
    for (int t = 0; t < 512; ++t) {
        if ((t & 3) == 0 && t + 32 < 512)
            __builtin_prefetch(&c[(size_t)(t + 32) * 128], 0, 1);
        mem = LIF_BETA * mem + c[(size_t)t * 128];
        float sv = (mem - LIF_THRESH) > 0.f ? 1.f : 0.f;
        sum += sv;
        mem -= sv * LIF_THRESH;
    }
    meanout[idx] = sum * (1.f / 512.f);
}

// ---------------------------------------------------------------- readout (tiny)
__global__ void readout(const float* __restrict__ mean, const float* __restrict__ Wr,
                        const float* __restrict__ br, float* __restrict__ out) {
    int idx = blockIdx.x * blockDim.x + threadIdx.x;
    if (idx < 256 * 36) {
        int b = idx / 36, o = idx - b * 36;
        const float* m = mean + b * 128;
        const float* w = Wr + o * 128;
        float s = 0.f;
#pragma unroll 4
        for (int k = 0; k < 128; ++k) s += m[k] * w[k];
        out[idx] = s + br[o];
    } else if (idx == 256 * 36) {
        out[idx] = 0.f;   // reference returns (logits, scalar 0)
    }
}

// ---------------------------------------------------------------- launcher
extern "C" void kernel_launch(void* const* d_in, const int* in_sizes, int n_in,
                              void* d_out, int out_size, void* d_ws, size_t ws_size,
                              hipStream_t stream) {
    const float* x     = (const float*)d_in[0];
    const float* Wi    = (const float*)d_in[1];
    const float* bi    = (const float*)d_in[2];
    const float* gamma = (const float*)d_in[3];
    const float* betaL = (const float*)d_in[4];
    const float* W1    = (const float*)d_in[5];
    const float* b1    = (const float*)d_in[6];
    const float* W2    = (const float*)d_in[7];
    const float* b2    = (const float*)d_in[8];
    const float* Wr    = (const float*)d_in[9];
    const float* br    = (const float*)d_in[10];

    // Workspace layout (all offsets 256B aligned), total ~101 MB:
    char* ws = (char*)d_ws;
    _Float16* WiH   = (_Float16*)(ws + 0);           // 128*480 f16   = 122880 B
    _Float16* W1H   = (_Float16*)(ws + 122880);      // 128*128 f16   =  32768 B
    _Float16* W2H   = (_Float16*)(ws + 155648);      // 128*128 f16   =  32768 B
    float*    meanb = (float*)   (ws + 188416);      // 256*128 f32   = 131072 B
    _Float16* Hbuf  = (_Float16*)(ws + 319488);      // 131072*128 f16 = 33.5 MB (H, then s1)
    float*    cur   = (float*)   (ws + 33873920);    // 131072*128 f32 = 67 MB   (cur1, then cur2)

    // 1) weights to f16
    cvt_f32_to_f16<<<(61440 + 255) / 256, 256, 0, stream>>>(Wi, WiH, 61440);
    cvt_f32_to_f16<<<(16384 + 255) / 256, 256, 0, stream>>>(W1, W1H, 16384);
    cvt_f32_to_f16<<<(16384 + 255) / 256, 256, 0, stream>>>(W2, W2H, 16384);

    // 2) fused transpose + input GEMM + LayerNorm + tanh -> Hbuf (f16)
    gemm_in_ln<<<8192, 256, 0, stream>>>(x, WiH, bi, gamma, betaL, Hbuf);

    // 3) cur1 = H @ W1^T + b1 ; LIF layer 1 -> spikes s1 (reuse Hbuf)
    gemm_hidden<<<4096, 256, 0, stream>>>(Hbuf, W1H, b1, cur);
    lif_spk<<<128, 256, 0, stream>>>(cur, Hbuf);

    // 4) cur2 = s1 @ W2^T + b2 ; LIF layer 2 with time-mean
    gemm_hidden<<<4096, 256, 0, stream>>>(Hbuf, W2H, b2, cur);
    lif_mean<<<128, 256, 0, stream>>>(cur, meanb);

    // 5) logits + trailing scalar 0
    readout<<<(9217 + 255) / 256, 256, 0, stream>>>(meanb, Wr, br, (float*)d_out);
}
